// SruEmb_6631429505568
// MI455X (gfx1250) — matrix operand order repaired and checked by reading, not compiled, not run on hardware
//
#include <hip/hip_runtime.h>
#include <hip/hip_bf16.h>
#include <cstdint>
#include <cstddef>

// Problem constants (from reference): B=16, T=2048, D=1024, L=4
#define BATCH 16
#define SEQ   2048
#define DIM   1024
#define NDIM  3072            // 3*D
#define TBROWS 32768          // T*B
#define LAYERS 4

typedef __attribute__((ext_vector_type(16))) __bf16 v16bf;
typedef __attribute__((ext_vector_type(8)))  __bf16 v8bf;
typedef __attribute__((ext_vector_type(8)))  float  v8f;

__device__ __forceinline__ unsigned short f2bf(float f) {
  union { float f; unsigned int u; } v; v.f = f;
  unsigned int r = 0x7FFFu + ((v.u >> 16) & 1u);   // round-to-nearest-even
  return (unsigned short)((v.u + r) >> 16);
}
__device__ __forceinline__ float bf2f(unsigned short h) {
  union { unsigned int u; float f; } v; v.u = ((unsigned int)h) << 16;
  return v.f;
}

// --------------------------------------------------------------------------
// CDNA5 async global->LDS path (ASYNCcnt-tracked), guarded so the file
// compiles on toolchains that lack the builtin (falls back to sync staging).
// Builtin signature (from hipcc diagnostic): params are pointers to
// int __attribute__((vector_size(16))), global side in AS(1), LDS in AS(3).
// --------------------------------------------------------------------------
#if defined(__HIP_DEVICE_COMPILE__) && defined(__gfx1250__) && \
    __has_builtin(__builtin_amdgcn_global_load_async_to_lds_b128)
#define USE_ASYNC_LDS 1
#else
#define USE_ASYNC_LDS 0
#endif

#if USE_ASYNC_LDS
typedef int v4i_vec __attribute__((vector_size(16)));
typedef __attribute__((address_space(1))) v4i_vec* gptr_b128;
typedef __attribute__((address_space(3))) v4i_vec* lptr_b128;

__device__ __forceinline__ void cp32_async(const unsigned short* g, unsigned short* l) {
  // copy 32 bytes (16 bf16) global -> LDS as two async b128 transfers
  __builtin_amdgcn_global_load_async_to_lds_b128((gptr_b128)g, (lptr_b128)l, 0, 0);
  __builtin_amdgcn_global_load_async_to_lds_b128((gptr_b128)(g + 8), (lptr_b128)(l + 8), 0, 0);
}
template <int N>
__device__ __forceinline__ void wait_asynccnt() {
#if __has_builtin(__builtin_amdgcn_s_wait_asynccnt)
  __builtin_amdgcn_s_wait_asynccnt(N);
#else
  asm volatile("s_wait_asynccnt %0" ::"i"(N) : "memory");
#endif
}
#endif

// ---------------------------------------------------------------------------
// Kernel 0: W [L, D, 3D] f32  ->  Wt [L, 3D, D] bf16 (transposed, per layer)
// Runs once; hoists the fp32->bf16 convert + transpose out of the GEMM loop.
// ---------------------------------------------------------------------------
__global__ __launch_bounds__(256) void transpose_w_kernel(
    const float* __restrict__ W, unsigned short* __restrict__ Wt) {
  __shared__ unsigned short tile[32][33];
  const int l  = blockIdx.z;
  const int n0 = blockIdx.x * 32;
  const int k0 = blockIdx.y * 32;
  const int tx = threadIdx.x & 31;
  const int ty = threadIdx.x >> 5;        // 0..7
  const float* Wl = W + (size_t)l * DIM * NDIM;
  unsigned short* Wtl = Wt + (size_t)l * NDIM * DIM;
#pragma unroll
  for (int i = 0; i < 4; ++i) {
    int k = k0 + ty + i * 8;
    tile[ty + i * 8][tx] = f2bf(Wl[(size_t)k * NDIM + n0 + tx]);  // [k][n]
  }
  __syncthreads();
#pragma unroll
  for (int i = 0; i < 4; ++i) {
    int n = n0 + ty + i * 8;
    Wtl[(size_t)n * DIM + k0 + tx] = tile[tx][ty + i * 8];        // Wt[n][k]
  }
}

// ---------------------------------------------------------------------------
// Kernel 1: x [B, T, D] fp32  ->  h [T*B, D] bf16   (time-major for the scan)
// ---------------------------------------------------------------------------
__global__ __launch_bounds__(256) void convert_x_kernel(
    const float* __restrict__ x, unsigned short* __restrict__ h) {
  size_t i = (size_t)blockIdx.x * 256 + threadIdx.x;   // b-major flat index
  int d = (int)(i & (DIM - 1));
  size_t r = i >> 10;              // = b*SEQ + t
  int t = (int)(r & (SEQ - 1));
  int b = (int)(r >> 11);
  h[((size_t)(t * BATCH + b)) * DIM + d] = f2bf(x[i]);
}

// ---------------------------------------------------------------------------
// Kernel 2: U = A * Wt^T  (A: [TBROWS, DIM] bf16, Wt: [NDIM, DIM] bf16)
// Block tile 128x128, K-step 32, 256 threads = 8 waves in a 2(m) x 4(n) grid,
// each wave owns a 64x32 sub-tile = 4x2 WMMA 16x16 accumulators.
// Double-buffered LDS; async global->LDS when available.
// ---------------------------------------------------------------------------
__global__ __launch_bounds__(256) void sru_gemm_bf16(
    const unsigned short* __restrict__ A,
    const unsigned short* __restrict__ Wt,
    unsigned short* __restrict__ U) {
  // [buf][mat(0=A,1=B)][row][k], 40 KB total; reused as C staging at epilogue
  __shared__ unsigned short smem[2][2][128][40];

  const int tid  = threadIdx.x;
  const int lane = tid & 31;               // wave32 (gfx1250)
  const int wid  = tid >> 5;
  const int wm   = wid & 1;                // 0..1  (64 rows each)
  const int wn   = wid >> 1;               // 0..3  (32 cols each)
  const size_t blockM = (size_t)blockIdx.y * 128;
  const size_t blockN = (size_t)blockIdx.x * 128;

  v8f acc[4][2];
#pragma unroll
  for (int mt = 0; mt < 4; ++mt)
#pragma unroll
    for (int nt = 0; nt < 2; ++nt) acc[mt][nt] = {};

  const int arow = tid >> 1;               // 0..127
  const int acol = (tid & 1) << 4;         // 0 or 16
  const unsigned short* gA = A + (blockM + arow) * DIM + acol;
  const unsigned short* gB = Wt + (blockN + arow) * DIM + acol;

#if USE_ASYNC_LDS
  // prologue: issue k-tile 0 into buffer 0
  cp32_async(gA, &smem[0][0][arow][acol]);
  cp32_async(gB, &smem[0][1][arow][acol]);
#endif

  for (int ko = 0; ko < DIM / 32; ++ko) {
    const int buf = ko & 1;
#if USE_ASYNC_LDS
    if (ko < DIM / 32 - 1) {
      const int k1 = (ko + 1) << 5;
      cp32_async(gA + k1, &smem[buf ^ 1][0][arow][acol]);
      cp32_async(gB + k1, &smem[buf ^ 1][1][arow][acol]);
      wait_asynccnt<4>();   // tile ko complete; tile ko+1 still in flight
    } else {
      wait_asynccnt<0>();
    }
#else
    {
      const int k0 = ko << 5;
      *(uint4*)&smem[buf][0][arow][acol]     = *(const uint4*)(gA + k0);
      *(uint4*)&smem[buf][0][arow][acol + 8] = *(const uint4*)(gA + k0 + 8);
      *(uint4*)&smem[buf][1][arow][acol]     = *(const uint4*)(gB + k0);
      *(uint4*)&smem[buf][1][arow][acol + 8] = *(const uint4*)(gB + k0 + 8);
      if (ko < DIM / 32 - 1) {
        __builtin_prefetch(gA + k0 + 32, 0, 3);
        __builtin_prefetch(gB + k0 + 32, 0, 3);
      }
    }
#endif
    __syncthreads();

    // --- B fragments (32x16): lane n<16 holds K{0-7,16-23} of col n, lane n+16
    //     holds K{8-15,24-31} (transposed LDS) ---
    v16bf bfrag[2];
#pragma unroll
    for (int nt = 0; nt < 2; ++nt) {
      const unsigned short* p = &smem[buf][1][wn * 32 + nt * 16 + (lane & 15)][0];
      const int kb = (lane < 16) ? 0 : 8;
      v8bf lo = *(const v8bf*)(p + kb);
      v8bf hi = *(const v8bf*)(p + kb + 16);
#pragma unroll
      for (int i = 0; i < 8; ++i) { bfrag[nt][i] = lo[i]; bfrag[nt][8 + i] = hi[i]; }
    }
    // --- A fragments (16x32): lane m<16 holds K{0-7,16-23}, lane m+16 K{8-15,24-31}
#pragma unroll
    for (int mt = 0; mt < 4; ++mt) {
      const unsigned short* p = &smem[buf][0][wm * 64 + mt * 16 + (lane & 15)][0];
      const int kb = (lane < 16) ? 0 : 8;
      v8bf lo = *(const v8bf*)(p + kb);
      v8bf hi = *(const v8bf*)(p + kb + 16);
      v16bf afrag;
#pragma unroll
      for (int i = 0; i < 8; ++i) { afrag[i] = lo[i]; afrag[8 + i] = hi[i]; }
#pragma unroll
      for (int nt = 0; nt < 2; ++nt) {
        acc[mt][nt] = __builtin_amdgcn_wmma_f32_16x16x32_bf16(
            false, afrag, false, bfrag[nt], (short)0, acc[mt][nt], false, false);
      }
    }
    __syncthreads();
  }

  // --- epilogue: stage C tile in LDS (reusing smem), then coalesced b128 out.
  // C/D layout: VGPR j -> row j (lanes 0-15) / row 8+j (lanes 16-31).
  unsigned short* Cs = &smem[0][0][0][0];   // viewed as [128][136]
#pragma unroll
  for (int mt = 0; mt < 4; ++mt)
#pragma unroll
    for (int nt = 0; nt < 2; ++nt) {
      const int m0 = wm * 64 + mt * 16 + ((lane >> 4) << 3);
      const int n0 = wn * 32 + nt * 16 + (lane & 15);
#pragma unroll
      for (int j = 0; j < 8; ++j)
        Cs[(size_t)(m0 + j) * 136 + n0] = f2bf(acc[mt][nt][j]);
    }
  __syncthreads();
  {
    const int m    = tid >> 1;
    const int half = tid & 1;          // 64-column halves
    const uint4* src = (const uint4*)&Cs[(size_t)m * 136 + half * 64];
    unsigned short* dstp = U + (blockM + m) * (size_t)NDIM + blockN + half * 64;
#pragma unroll
    for (int i = 0; i < 8; ++i) ((uint4*)dstp)[i] = src[i];   // 128 B contiguous
  }
}

// ---------------------------------------------------------------------------
// Kernel 3: sequential SRU scan over T; one thread per (b, d) channel.
// c_t = f*c + (1-f)*x~ ; out = r*tanh(c) + (1-r)*h_in
// ---------------------------------------------------------------------------
__global__ __launch_bounds__(256) void sru_recurrence(
    const unsigned short* __restrict__ U,     // [T*B, 3D] bf16
    const float* __restrict__ bias,           // [2D]
    const unsigned short* __restrict__ h_in,  // [T*B, D] bf16
    unsigned short* __restrict__ h_out) {     // [T*B, D] bf16
  const int idx = blockIdx.x * 256 + threadIdx.x;  // 0..B*D-1
  const int d = idx & (DIM - 1);
  const int b = idx >> 10;
  const float bf = bias[d];
  const float br = bias[DIM + d];
  float c = 0.0f;
  for (int t = 0; t < SEQ; ++t) {
    const size_t row = (size_t)(t * BATCH + b);
    const unsigned short* u = U + row * NDIM;
    const float xt = bf2f(u[d]);
    const float zf = bf2f(u[DIM + d]);
    const float zr = bf2f(u[2 * DIM + d]);
    const float f = 1.0f / (1.0f + __expf(-(zf + bf)));
    const float r = 1.0f / (1.0f + __expf(-(zr + br)));
    c = f * c + (1.0f - f) * xt;
    const float h = bf2f(h_in[row * DIM + d]);
    h_out[row * DIM + d] = f2bf(r * tanhf(c) + (1.0f - r) * h);
  }
}

// ---------------------------------------------------------------------------
// Kernel 4: out[b, :] = h[(lengths[b]-1)*B + b, :]  (bf16 -> f32)
// ---------------------------------------------------------------------------
__global__ __launch_bounds__(256) void select_last_kernel(
    const unsigned short* __restrict__ h, const int* __restrict__ lengths,
    float* __restrict__ out) {
  const int idx = blockIdx.x * 256 + threadIdx.x;  // 0..B*D-1
  const int d = idx & (DIM - 1);
  const int b = idx >> 10;
  const int t = lengths[b] - 1;
  out[(size_t)b * DIM + d] = bf2f(h[((size_t)(t * BATCH + b)) * DIM + d]);
}

// ---------------------------------------------------------------------------
extern "C" void kernel_launch(void* const* d_in, const int* in_sizes, int n_in,
                              void* d_out, int out_size, void* d_ws, size_t ws_size,
                              hipStream_t stream) {
  (void)in_sizes; (void)n_in; (void)out_size; (void)ws_size;
  const float* x       = (const float*)d_in[0];  // [B, T, D]
  const int*   lengths = (const int*)d_in[1];    // [B]
  const float* W       = (const float*)d_in[2];  // [L, D, 3D]
  const float* bias    = (const float*)d_in[3];  // [L, 2D]
  float* out = (float*)d_out;                    // [B, D]

  char* ws = (char*)d_ws;
  const size_t HBYTES = (size_t)TBROWS * DIM * 2;            // 64 MB
  unsigned short* h0 = (unsigned short*)ws;
  unsigned short* h1 = (unsigned short*)(ws + HBYTES);
  unsigned short* U  = (unsigned short*)(ws + 2 * HBYTES);   // 192 MB
  unsigned short* Wt = (unsigned short*)(ws + 2 * HBYTES + (size_t)TBROWS * NDIM * 2);  // 24 MB

  transpose_w_kernel<<<dim3(NDIM / 32, DIM / 32, LAYERS), dim3(256), 0, stream>>>(W, Wt);

  {
    const size_t total = (size_t)BATCH * SEQ * DIM;   // 33,554,432
    convert_x_kernel<<<dim3((unsigned)(total / 256)), dim3(256), 0, stream>>>(x, h0);
  }

  unsigned short* cur = h0;
  unsigned short* nxt = h1;
  for (int l = 0; l < LAYERS; ++l) {
    sru_gemm_bf16<<<dim3(NDIM / 128, TBROWS / 128), dim3(256), 0, stream>>>(
        cur, Wt + (size_t)l * NDIM * DIM, U);
    sru_recurrence<<<dim3((BATCH * DIM) / 256), dim3(256), 0, stream>>>(
        U, bias + (size_t)l * 2 * DIM, cur, nxt);
    unsigned short* tmp = cur; cur = nxt; nxt = tmp;
  }
  select_last_kernel<<<dim3((BATCH * DIM) / 256), dim3(256), 0, stream>>>(cur, lengths, out);
}
